// VanillaRNN_37726992728097
// MI455X (gfx1250) — matrix operand order repaired and checked by reading, not compiled, not run on hardware
//
#include <hip/hip_runtime.h>

#define BATCH 512
#define SEQ   128
#define HID   512
#define NOUT  10

typedef __attribute__((ext_vector_type(16))) __bf16 v16bf;
typedef __attribute__((ext_vector_type(8)))  float  v8f;

union FragU {
  uint4 q[2];
  v16bf v;
};

__device__ __forceinline__ unsigned short f2bf(float f) {
  unsigned int u = __float_as_uint(f);
  u += 0x7FFFu + ((u >> 16) & 1u);   // round-to-nearest-even
  return (unsigned short)(u >> 16);
}
__device__ __forceinline__ float bf2f(unsigned short s) {
  return __uint_as_float(((unsigned int)s) << 16);
}

// One-time per launch: WhhT[n][k] = bf16(Whh[k][n]); h0 = 0.
__global__ void rnn_prep_kernel(const float* __restrict__ Whh,
                                unsigned short* __restrict__ WhhT,
                                unsigned short* __restrict__ h0) {
  int idx = blockIdx.x * blockDim.x + threadIdx.x;
  if (idx < HID * HID) {
    int k = idx >> 9;          // row of Whh
    int n = idx & (HID - 1);   // col of Whh
    WhhT[n * HID + k] = f2bf(Whh[idx]);
    h0[idx] = 0;               // bf16 zero
  }
}

// One RNN step: hout = tanh(hin @ Whh + x[:,t] * Whx + bh), bf16 in/out, f32 accum.
// Grid (8,8), block 128 (4 waves). Wave tile: 32(M=batch) x 32(N=hidden).
__global__ void __launch_bounds__(128)
rnn_step_kernel(const unsigned short* __restrict__ hin,   // [512,512] bf16 row-major
                unsigned short* __restrict__ hout,        // [512,512] bf16 row-major
                const unsigned short* __restrict__ WhhT,  // [N=512,K=512] bf16 (Whh transposed)
                const float* __restrict__ x,              // [B,T] f32
                const float* __restrict__ Whx,            // [H] f32
                const float* __restrict__ bh,             // [H] f32
                int t) {
  const int lane  = threadIdx.x & 31;
  const int wave  = threadIdx.x >> 5;
  const int waveM = wave & 1;
  const int waveN = wave >> 1;
  const int m0 = blockIdx.y * 64 + waveM * 32;   // batch-tile origin
  const int n0 = blockIdx.x * 64 + waveN * 32;   // hidden-tile origin
  const int l15 = lane & 15;
  const int lhi = lane >> 4;                      // 0 or 1 (half-wave)

  // A (16-bit 16x32): lane<16 holds K {0..7,16..23}; lane>=16 holds K {8..15,24..31}
  // B (16-bit 32x16): lane<16 holds K {0..7,16..23}; lane>=16 holds K {8..15,24..31}
  const unsigned short* arow0 = hin  + (m0 + l15) * HID;
  const unsigned short* arow1 = arow0 + 16 * HID;
  const unsigned short* brow0 = WhhT + (n0 + l15) * HID;
  const unsigned short* brow1 = brow0 + 16 * HID;
  const int acol = lhi * 8;    // second quad at +16
  const int bcol = lhi * 8;    // second quad at +16

  v8f acc[2][2];
  acc[0][0] = (v8f){}; acc[0][1] = (v8f){};
  acc[1][0] = (v8f){}; acc[1][1] = (v8f){};

  for (int k = 0; k < HID; k += 32) {
    const unsigned short* pa0 = arow0 + k + acol;
    const unsigned short* pa1 = arow1 + k + acol;
    const unsigned short* pb0 = brow0 + k + bcol;
    const unsigned short* pb1 = brow1 + k + bcol;

    if (k + 32 < HID) {
      // locality=3 -> near-cache (WGP$) policy: data is L2-resident, stage it
      // into the WGP cache for the next K block.
      __builtin_prefetch(pa0 + 32, 0, 3);
      __builtin_prefetch(pb0 + 32, 0, 3);
    }

    FragU a0, a1, b0, b1;
    a0.q[0] = *(const uint4*)(pa0);      a0.q[1] = *(const uint4*)(pa0 + 16);
    a1.q[0] = *(const uint4*)(pa1);      a1.q[1] = *(const uint4*)(pa1 + 16);
    b0.q[0] = *(const uint4*)(pb0);      b0.q[1] = *(const uint4*)(pb0 + 16);
    b1.q[0] = *(const uint4*)(pb1);      b1.q[1] = *(const uint4*)(pb1 + 16);

    acc[0][0] = __builtin_amdgcn_wmma_f32_16x16x32_bf16(false, a0.v, false, b0.v,
                                                        (short)0, acc[0][0], false, false);
    acc[0][1] = __builtin_amdgcn_wmma_f32_16x16x32_bf16(false, a0.v, false, b1.v,
                                                        (short)0, acc[0][1], false, false);
    acc[1][0] = __builtin_amdgcn_wmma_f32_16x16x32_bf16(false, a1.v, false, b0.v,
                                                        (short)0, acc[1][0], false, false);
    acc[1][1] = __builtin_amdgcn_wmma_f32_16x16x32_bf16(false, a1.v, false, b1.v,
                                                        (short)0, acc[1][1], false, false);
  }

  // Epilogue. C/D layout: VGPR r, lane<16 -> (M=r,   N=lane);
  //                       lane>=16      -> (M=r+8, N=lane-16)
  const int rbase = lhi * 8;
#pragma unroll
  for (int im = 0; im < 2; ++im) {
#pragma unroll
    for (int in_ = 0; in_ < 2; ++in_) {
      const int colg = n0 + in_ * 16 + l15;
      const float whx = Whx[colg];
      const float bhv = bh[colg];
#pragma unroll
      for (int r = 0; r < 8; ++r) {
        const int rowg = m0 + im * 16 + rbase + r;
        const float xv  = x[rowg * SEQ + t];
        const float pre = acc[im][in_][r] + xv * whx + bhv;
        hout[rowg * HID + colg] = f2bf(tanhf(pre));
      }
    }
  }
}

// Final projection: out[b][o] = sum_k h[b][k] * Wph[k][o] + bo[o]  (512x10, tiny)
__global__ void rnn_proj_kernel(const unsigned short* __restrict__ h,
                                const float* __restrict__ Wph,
                                const float* __restrict__ bo,
                                float* __restrict__ out) {
  int idx = blockIdx.x * blockDim.x + threadIdx.x;
  if (idx >= BATCH * NOUT) return;
  int b = idx / NOUT;
  int o = idx - b * NOUT;
  float acc = bo[o];
  for (int k = 0; k < HID; ++k)
    acc += bf2f(h[b * HID + k]) * Wph[k * NOUT + o];
  out[idx] = acc;
}

extern "C" void kernel_launch(void* const* d_in, const int* in_sizes, int n_in,
                              void* d_out, int out_size, void* d_ws, size_t ws_size,
                              hipStream_t stream) {
  const float* x   = (const float*)d_in[0];  // [512,128]
  const float* Whx = (const float*)d_in[1];  // [1,512]
  const float* Whh = (const float*)d_in[2];  // [512,512]
  const float* Wph = (const float*)d_in[3];  // [512,10]
  const float* bh  = (const float*)d_in[4];  // [1,512]
  const float* bo  = (const float*)d_in[5];  // [1,10]

  // Workspace layout (1.5 MB): WhhT bf16 | hA bf16 | hB bf16
  unsigned short* WhhT = (unsigned short*)d_ws;
  unsigned short* hA   = WhhT + HID * HID;
  unsigned short* hB   = hA   + BATCH * HID;

  rnn_prep_kernel<<<dim3((HID * HID + 255) / 256), dim3(256), 0, stream>>>(Whh, WhhT, hA);

  for (int t = 0; t < SEQ; ++t) {
    const unsigned short* hin = (t & 1) ? hB : hA;
    unsigned short*       hout = (t & 1) ? hA : hB;
    rnn_step_kernel<<<dim3(HID / 64, BATCH / 64), dim3(128), 0, stream>>>(
        hin, hout, WhhT, x, Whx, bh, t);
  }

  // SEQ=128 is even: last step (t=127, odd) wrote hA.
  rnn_proj_kernel<<<dim3((BATCH * NOUT + 255) / 256), dim3(256), 0, stream>>>(
      hA, Wph, bo, (float*)d_out);
}
